// MLACausalAttention_2680059593235
// MI455X (gfx1250) — matrix operand order repaired and checked by reading, not compiled, not run on hardware
//
#include <hip/hip_runtime.h>
#include <hip/hip_bf16.h>

// MLA causal attention for MI455X (gfx1250, wave32).
// - all matrix math on v_wmma_f32_16x16x32_bf16 (f32 accumulate)
// - all fragment loads are contiguous 2 x b128 (operands pre-packed into
//   k-contiguous fragment order, k = 16g + s)
// - global->LDS staging uses gfx1250 async-to-LDS DMA (ASYNCcnt) with
//   double-buffered tiles: compute on tile t overlaps DMA of tile t+1.

typedef __attribute__((ext_vector_type(16))) __bf16 v16bf;
typedef __attribute__((ext_vector_type(8)))  float  v8f;

typedef unsigned short u16;
typedef unsigned int   u32;

#define DEV __device__ __forceinline__

union Frag16 {
    v16bf v;
    u16   u[16];
    u32   d[8];
};

DEV u16 f2bf(float x) {               // round-to-nearest-even f32 -> bf16
    u32 u = __float_as_uint(x);
    u += 0x7fffu + ((u >> 16) & 1u);
    return (u16)(u >> 16);
}
DEV float bf2f(u16 h) { return __uint_as_float(((u32)h) << 16); }

DEV v8f vzero() {
    v8f z;
#pragma unroll
    for (int r = 0; r < 8; ++r) z[r] = 0.0f;
    return z;
}

DEV v8f wmma_bf16(const Frag16& a, const Frag16& b, v8f c) {
    return __builtin_amdgcn_wmma_f32_16x16x32_bf16(
        false, a.v, false, b.v, (short)0, c, false, false);
}

// A-fragment (16x32): halves 0-15 -> K = 16g..16g+15
DEV void load_frag_a(Frag16& f, const u16* rowbase, int g) {
    *(uint4*)&f.d[0] = *(const uint4*)(rowbase + 16 * g);
    *(uint4*)&f.d[4] = *(const uint4*)(rowbase + 16 * g + 8);
}
// B-fragment (32x16): halves h -> K = h + 16g; p points at 16 contiguous K values
DEV void load_frag_b(Frag16& f, const u16* p) {
    *(uint4*)&f.d[0] = *(const uint4*)(p);
    *(uint4*)&f.d[4] = *(const uint4*)(p + 8);
}

// gfx1250 async global->LDS 16B DMA, per-lane addresses (GV mode).
// ldsoff = low 32 bits of flat shared pointer == LDS byte offset.
DEV void async_b128(u32 ldsoff, const void* gaddr) {
    asm volatile("global_load_async_to_lds_b128 %0, %1, off"
                 :: "v"(ldsoff), "v"(gaddr) : "memory");
}
DEV u32 lds_off(const void* p) { return (u32)(size_t)p; }

// ---------------------------------------------------------------------------
// fp32 -> bf16 conversion (row-major, for A operands)
// ---------------------------------------------------------------------------
__global__ void cvt_f32_bf16(const float* __restrict__ in,
                             u16* __restrict__ out, int n) {
    int i = blockIdx.x * blockDim.x + threadIdx.x;
    if (i < n) out[i] = f2bf(in[i]);
}

// ---------------------------------------------------------------------------
// fp32 [K,N] -> bf16 packed B-fragment order: out[((k>>5)*N + n)*32 + (k&31)]
// ---------------------------------------------------------------------------
__global__ void cvt_pack_b(const float* __restrict__ in,
                           u16* __restrict__ out, int K, int N) {
    int i = blockIdx.x * blockDim.x + threadIdx.x;
    if (i >= K * N) return;
    int k = i / N, n = i - k * N;
    out[(((size_t)(k >> 5) * N + n) << 5) + (k & 31)] = f2bf(in[i]);
}

// ---------------------------------------------------------------------------
// Pack v (bf16 token-major [B*L, NH*128]) into per-(b, kv-tile, head)
// B-fragment order: vp[(((b*64 + (l>>5))*16 + h)*128 + d)*32 + (l&31)]
// ---------------------------------------------------------------------------
__global__ void pack_v(const u16* __restrict__ in, u16* __restrict__ out) {
    int i = blockIdx.x * blockDim.x + threadIdx.x;   // < 4096*2048
    int bl = i >> 11;
    int c  = i & 2047;
    int h  = c >> 7, d = c & 127;
    int b  = bl >> 11, l = bl & 2047;
    size_t o = ((((size_t)b * 64 + (l >> 5)) * 16 + h) * 128 + d) * 32 + (l & 31);
    out[o] = in[i];
}

// ---------------------------------------------------------------------------
// Tiled bf16 GEMM: C[M,N] = A[M,K] * Bp (packed), f32 accum.
// Block tile 64(M) x 256(N), K-step 32, double-buffered async LDS staging.
// 8 waves = 2(M) x 4(N); each wave: 32x64 = eight 16x16x32 WMMAs per K-step.
// ---------------------------------------------------------------------------
__global__ __launch_bounds__(256) void gemm_bf16(
    const u16* __restrict__ A, const u16* __restrict__ Bp,
    void* __restrict__ Cv, int M, int N, int K, int storeBF16)
{
    __shared__ alignas(16) u16 As[2][64 * 32];    //  2 x 4 KB, row-major
    __shared__ alignas(16) u16 Bs[2][256 * 32];   //  2 x 16 KB, frag order [n][k]

    const int tid  = threadIdx.x;
    const int lane = tid & 31, wave = tid >> 5;
    const int g = lane >> 4, ln = lane & 15;
    const int m0 = blockIdx.y * 64;
    const int n0 = blockIdx.x * 256;
    const int mg = (wave >> 2) * 32;   // 0 or 32
    const int ng = (wave & 3) * 64;    // 0,64,128,192
    const int ntile = K >> 5;

    // issue the 5 async-to-LDS DMAs for K-tile t into buffer bufI
    auto issue = [&](int t, int bufI) {
        int k0 = t << 5;
        {   // A tile 64x32 (16B per thread)
            int idx = tid * 8;
            int ai = idx >> 5, ak = idx & 31;
            async_b128(lds_off(&As[bufI][idx]),
                       A + (size_t)(m0 + ai) * K + k0 + ak);
        }
        {   // B tile: contiguous 16 KB chunk of packed B (64B per thread)
            const u16* src = Bp + (((size_t)t * N + n0) << 5) + tid * 32;
            u32 dst = lds_off(&Bs[bufI][tid * 32]);
#pragma unroll
            for (int q = 0; q < 4; ++q)
                async_b128(dst + q * 16, src + q * 8);
        }
    };

    v8f acc[2][4];
#pragma unroll
    for (int i = 0; i < 2; ++i)
#pragma unroll
        for (int j = 0; j < 4; ++j) acc[i][j] = vzero();

    issue(0, 0);
    for (int t = 0; t < ntile; ++t) {
        const int cur = t & 1;
        if (t + 1 < ntile) {
            issue(t + 1, cur ^ 1);
            // async loads complete in order: <=5 outstanding => tile t landed
            asm volatile("s_wait_asynccnt 0x5" ::: "memory");
        } else {
            asm volatile("s_wait_asynccnt 0x0" ::: "memory");
        }
        __syncthreads();

        Frag16 fa[2];
#pragma unroll
        for (int i = 0; i < 2; ++i)
            load_frag_a(fa[i], &As[cur][(mg + i * 16 + ln) * 32], g);
#pragma unroll
        for (int j = 0; j < 4; ++j) {
            Frag16 fb;
            load_frag_b(fb, &Bs[cur][(ng + j * 16 + ln) * 32 + 16 * g]);
#pragma unroll
            for (int i = 0; i < 2; ++i)
                acc[i][j] = wmma_bf16(fa[i], fb, acc[i][j]);
        }
        __syncthreads();
    }

    // store: C/D layout: VGPR r, half-wave g -> row m = r+8g, col = lane%16
#pragma unroll
    for (int i = 0; i < 2; ++i)
#pragma unroll
        for (int j = 0; j < 4; ++j) {
            int coln = n0 + ng + j * 16 + ln;
#pragma unroll
            for (int r = 0; r < 8; ++r) {
                int rowm = m0 + mg + i * 16 + r + 8 * g;
                if (storeBF16)
                    ((u16*)Cv)[(size_t)rowm * N + coln] = f2bf(acc[i][j][r]);
                else
                    ((float*)Cv)[(size_t)rowm * N + coln] = acc[i][j][r];
            }
        }
}

// ---------------------------------------------------------------------------
// In-place RoPE on bf16 tensor laid out [B*L, NH*128].
// ---------------------------------------------------------------------------
__global__ void rope_inplace(u16* __restrict__ t) {
    int idx = blockIdx.x * blockDim.x + threadIdx.x;  // < 2*2048*16*64
    int i  = idx & 63;
    int h  = (idx >> 6) & 15;
    int bl = idx >> 10;             // b*L + l
    int l  = bl & 2047;
    float inv = __expf(-(float)(2 * i) * (9.210340371976184f / 128.0f));
    float ang = (float)l * inv;
    float s, c;
    __sincosf(ang, &s, &c);
    size_t base = (size_t)bl * 2048 + h * 128;
    float t1 = bf2f(t[base + i]);
    float t2 = bf2f(t[base + i + 64]);
    t[base + i]      = f2bf(t1 * c - t2 * s);
    t[base + i + 64] = f2bf(t2 * c + t1 * s);
}

// ---------------------------------------------------------------------------
// Flash attention (causal).  q/k bf16 token-major [B*L, NH*128]; v packed.
// Block = (b, h, 128 q-rows); 8 waves each own 16 q rows.
// K/V tiles double-buffered via async-to-LDS DMA.
// ---------------------------------------------------------------------------
__global__ __launch_bounds__(256) void mla_flash_attn(
    const u16* __restrict__ Q, const u16* __restrict__ Kb,
    const u16* __restrict__ Vp, u16* __restrict__ O)
{
    __shared__ alignas(16) u16 Ks[2][32 * 128];   // row-major [kv][d]
    __shared__ alignas(16) u16 Vs[2][128 * 32];   // fragment order [d][kv]
    __shared__ alignas(16) u16 Ps[8][16 * 32];    // per-wave P staging

    const int tid  = threadIdx.x;
    const int lane = tid & 31, wave = tid >> 5;
    const int g = lane >> 4, ln = lane & 15;
    const int q0 = blockIdx.x * 128;
    const int h  = blockIdx.y;
    const int b  = blockIdx.z;
    const int L = 2048, Dtok = 2048;
    const int qbase = q0 + wave * 16;
    const float scale = 0.08838834764831845f;  // 1/sqrt(128)
    const int ntile = (q0 >> 5) + 4;           // causal kv tiles of 32

    // issue the 4 async-to-LDS DMAs for kv-tile t into buffer bufI
    auto issue = [&](int t, int bufI) {
        int kv0 = t << 5;
        {   // K tile 32x128 row-major (32B per thread)
            int row = tid >> 3, col = (tid & 7) * 16;
            const u16* src = Kb + (size_t)(b * L + kv0 + row) * Dtok + h * 128 + col;
            u32 dst = lds_off(&Ks[bufI][row * 128 + col]);
            async_b128(dst,      src);
            async_b128(dst + 16, src + 8);
        }
        {   // V tile: contiguous packed 8 KB chunk (32B per thread)
            const u16* src = Vp + ((((size_t)b * 64 + t) * 16 + h) << 12) + tid * 16;
            u32 dst = lds_off(&Vs[bufI][tid * 16]);
            async_b128(dst,      src);
            async_b128(dst + 16, src + 8);
        }
    };

    // Register-resident Q fragments: 16 rows x 128 cols = 4 A-frags
    Frag16 qf[4];
    {
        const u16* qrow = Q + (size_t)(b * L + qbase + ln) * Dtok + h * 128;
#pragma unroll
        for (int dc = 0; dc < 4; ++dc)
            load_frag_a(qf[dc], qrow + dc * 32, g);
    }

    v8f o[8];
    float mrow[8], lrow[8];
#pragma unroll
    for (int j = 0; j < 8; ++j) o[j] = vzero();
#pragma unroll
    for (int r = 0; r < 8; ++r) { mrow[r] = -1e30f; lrow[r] = 0.0f; }

    issue(0, 0);
    for (int t = 0; t < ntile; ++t) {
        const int cur = t & 1;
        const int kv0 = t << 5;
        if (t + 1 < ntile) {
            issue(t + 1, cur ^ 1);
            asm volatile("s_wait_asynccnt 0x4" ::: "memory");
        } else {
            asm volatile("s_wait_asynccnt 0x0" ::: "memory");
        }
        __syncthreads();

        // S = Q K^T : two 16x16 tiles (kv cols 0-15 and 16-31)
        v8f s0 = vzero(), s1 = vzero();
#pragma unroll
        for (int dc = 0; dc < 4; ++dc) {
            Frag16 fb0, fb1;
            load_frag_b(fb0, &Ks[cur][ln * 128        + dc * 32 + 16 * g]);
            load_frag_b(fb1, &Ks[cur][(16 + ln) * 128 + dc * 32 + 16 * g]);
            s0 = wmma_bf16(qf[dc], fb0, s0);
            s1 = wmma_bf16(qf[dc], fb1, s1);
        }

        // online softmax (row m = r + 8g lives in one half-wave)
#pragma unroll
        for (int r = 0; r < 8; ++r) {
            int qrow = qbase + r + 8 * g;
            float v0 = s0[r] * scale;
            float v1 = s1[r] * scale;
            if (kv0 + ln > qrow)      v0 = -1e30f;   // causal mask
            if (kv0 + 16 + ln > qrow) v1 = -1e30f;
            float mx = fmaxf(v0, v1);
#pragma unroll
            for (int off = 1; off < 16; off <<= 1)
                mx = fmaxf(mx, __shfl_xor(mx, off, 32));
            float mnew  = fmaxf(mrow[r], mx);
            float alpha = __expf(mrow[r] - mnew);
            float p0 = __expf(v0 - mnew);
            float p1 = __expf(v1 - mnew);
            float ps = p0 + p1;
#pragma unroll
            for (int off = 1; off < 16; off <<= 1)
                ps += __shfl_xor(ps, off, 32);
            lrow[r] = lrow[r] * alpha + ps;
            mrow[r] = mnew;
#pragma unroll
            for (int j = 0; j < 8; ++j) o[j][r] *= alpha;
            // stage P (C layout -> A layout) via per-wave LDS region;
            // same-wave LDS ops are in-order on CDNA5.
            Ps[wave][(r + 8 * g) * 32 + ln]      = f2bf(p0);
            Ps[wave][(r + 8 * g) * 32 + ln + 16] = f2bf(p1);
        }

        // O += P(16x32) * V(32x128)
        Frag16 pa;
        load_frag_a(pa, &Ps[wave][0] + ln * 32, g);
#pragma unroll
        for (int j = 0; j < 8; ++j) {
            Frag16 fv;
            load_frag_b(fv, &Vs[cur][(j * 16 + ln) * 32 + 16 * g]);
            o[j] = wmma_bf16(pa, fv, o[j]);
        }
        __syncthreads();
    }

    // normalize and store bf16 token-major
#pragma unroll
    for (int j = 0; j < 8; ++j) {
#pragma unroll
        for (int r = 0; r < 8; ++r) {
            size_t rowm = (size_t)(b * L + qbase + r + 8 * g);
            float val = o[j][r] / lrow[r];
            O[rowm * Dtok + h * 128 + j * 16 + ln] = f2bf(val);
        }
    }
}

// ---------------------------------------------------------------------------
// Host-side orchestration
// ---------------------------------------------------------------------------
extern "C" void kernel_launch(void* const* d_in, const int* in_sizes, int n_in,
                              void* d_out, int out_size, void* d_ws, size_t ws_size,
                              hipStream_t stream) {
    (void)in_sizes; (void)n_in; (void)out_size; (void)ws_size;
    const float* x     = (const float*)d_in[0];
    const float* W_DKV = (const float*)d_in[1];
    const float* W_UK  = (const float*)d_in[2];
    const float* W_UV  = (const float*)d_in[3];
    const float* W_DQ  = (const float*)d_in[4];
    const float* W_UQ  = (const float*)d_in[5];
    const float* W_out = (const float*)d_in[6];
    float* out = (float*)d_out;

    char* p = (char*)d_ws;
    auto alloc = [&](size_t elems) -> u16* {
        u16* r = (u16*)p;
        p += ((elems * 2 + 255) / 256) * 256;
        return r;
    };
    u16* xb   = alloc((size_t)4096 * 2048);
    u16* wdkv = alloc((size_t)2048 * 512);
    u16* wuk  = alloc((size_t)512  * 2048);
    u16* wuv  = alloc((size_t)512  * 2048);
    u16* wdq  = alloc((size_t)2048 * 1024);
    u16* wuq  = alloc((size_t)1024 * 2048);
    u16* wout = alloc((size_t)2048 * 2048);
    u16* ckv  = alloc((size_t)4096 * 512);
    u16* cq   = alloc((size_t)4096 * 1024);
    u16* kb   = alloc((size_t)4096 * 2048);
    u16* qb   = alloc((size_t)4096 * 2048);
    u16* vb   = alloc((size_t)4096 * 2048);
    u16* vp   = alloc((size_t)4096 * 2048);
    u16* ab   = alloc((size_t)4096 * 2048);

    cvt_f32_bf16<<<(4096 * 2048) / 256, 256, 0, stream>>>(x, xb, 4096 * 2048);
    auto pack = [&](const float* src, u16* dst, int K, int N) {
        cvt_pack_b<<<(K * N + 255) / 256, 256, 0, stream>>>(src, dst, K, N);
    };
    pack(W_DKV, wdkv, 2048, 512);
    pack(W_UK,  wuk,  512,  2048);
    pack(W_UV,  wuv,  512,  2048);
    pack(W_DQ,  wdq,  2048, 1024);
    pack(W_UQ,  wuq,  1024, 2048);
    pack(W_out, wout, 2048, 2048);

    auto gemm = [&](const u16* A, const u16* B, void* C,
                    int M, int N, int K, int bf) {
        dim3 grid(N / 256, M / 64);
        gemm_bf16<<<grid, 256, 0, stream>>>(A, B, C, M, N, K, bf);
    };
    gemm(xb,  wdkv, ckv, 4096, 512,  2048, 1);   // c_kv
    gemm(xb,  wdq,  cq,  4096, 1024, 2048, 1);   // c_q
    gemm(ckv, wuk,  kb,  4096, 2048, 512,  1);   // k (pre-rope)
    gemm(ckv, wuv,  vb,  4096, 2048, 512,  1);   // v
    gemm(cq,  wuq,  qb,  4096, 2048, 1024, 1);   // q (pre-rope)

    rope_inplace<<<(2 * 2048 * 16 * 64) / 256, 256, 0, stream>>>(qb);
    rope_inplace<<<(2 * 2048 * 16 * 64) / 256, 256, 0, stream>>>(kb);
    pack_v<<<(4096 * 2048) / 256, 256, 0, stream>>>(vb, vp);

    dim3 ag(2048 / 128, 16, 2);
    mla_flash_attn<<<ag, 256, 0, stream>>>(qb, kb, vp, ab);

    gemm(ab, wout, out, 4096, 2048, 2048, 0);    // final projection -> f32 out
}
